// MultiHeadAttion_27281632264678
// MI455X (gfx1250) — matrix operand order repaired and checked by reading, not compiled, not run on hardware
//
#include <hip/hip_runtime.h>
#include <hip/hip_bf16.h>

typedef __attribute__((ext_vector_type(16))) _Float16 v16h;
typedef __attribute__((ext_vector_type(8)))  _Float16 v8h;
typedef __attribute__((ext_vector_type(8)))  float    v8f;
typedef __attribute__((ext_vector_type(4)))  unsigned int u32x4;
typedef __attribute__((ext_vector_type(8)))  int      i32x8;
typedef __attribute__((ext_vector_type(4)))  int      i32x4;

#define NHEADS 4
#define HDIM   64
#define NB     8      // batch
#define CB     2      // channels
#define DD     2048   // depth / sequence
#define HW     256    // H*W
#define SC_STRIDE 2052  // padded LDS stride for scores (2052 % 64 == 4)
#define KROW      72    // K-stage LDS row stride in halves (TDM pad: 32 DW + 4 DW)

static __device__ __forceinline__ v8f wmma_f16(v16h a, v16h b, v8f c) {
    return __builtin_amdgcn_wmma_f32_16x16x32_f16(false, a, false, b, (short)0, c, false, false);
}

// TDM: async-DMA a [rows=16 x cols=64 halves] tile (row stride 256 halves in global)
// into LDS with a 4-DWORD pad every 32 DWORDs (-> LDS row stride 72 halves).
static __device__ __forceinline__ void tdm_load_k_tile(const _Float16* gsrc, unsigned int lds_off) {
    unsigned long long ga = (unsigned long long)(size_t)gsrc;
    u32x4 g0;
    g0[0] = 1u;                                   // count=1, user descriptor
    g0[1] = lds_off;                              // lds_addr (bytes)
    g0[2] = (unsigned int)ga;                     // global_addr lo
    g0[3] = (unsigned int)((ga >> 32) & 0x01FFFFFFu) | (2u << 30);  // addr hi | type=2
    i32x8 g1;
    g1[0] = 0x00010000                            // data_size = 2 bytes
          | 0x00100000                            // pad_enable
          | (4 << 22)                             // pad_interval: 2^(4+1)=32 DWORDs
          | (3 << 25);                            // pad_amount: 3+1 = 4 DWORDs
    g1[1] = (64 << 16);                           // tensor_dim0[15:0] = 64  (bits 63:48)
    g1[2] = (16 << 16);                           // tensor_dim1[15:0] = 16  (bits 95:80)
    g1[3] = (64 << 16);                           // tile_dim0 = 64          (bits 127:112)
    g1[4] = 16;                                   // tile_dim1 = 16, tile_dim2 = 0
    g1[5] = 256;                                  // tensor_dim0_stride = 256 elements
    g1[6] = 0;
    g1[7] = 0;
    i32x4 z4 = {0, 0, 0, 0};
#if __clang_major__ >= 23
    i32x8 z8 = {0, 0, 0, 0, 0, 0, 0, 0};
    __builtin_amdgcn_tensor_load_to_lds(g0, g1, z4, z4, z8, 0);
#else
    __builtin_amdgcn_tensor_load_to_lds(g0, g1, z4, z4, 0);
#endif
}

// ---------------- zero small accumulator buffer ----------------
__global__ void zero_kernel(float* p, int n) {
    int i = blockIdx.x * blockDim.x + threadIdx.x;
    if (i < n) p[i] = 0.0f;
}

// ---------------- 1) adaptive avg pool over H*W: wave per row ----------------
__global__ void pool_kernel(const float* __restrict__ x, float* __restrict__ xm) {
    int row  = (blockIdx.x * blockDim.x + threadIdx.x) >> 5;   // 0 .. N*C*D-1
    int lane = threadIdx.x & 31;
    const float* p = x + (size_t)row * HW;
    float s = 0.0f;
    for (int i = lane; i < HW; i += 32) s += __builtin_nontemporal_load(p + i);
    for (int m = 16; m; m >>= 1) s += __shfl_xor(s, m, 32);
    if (lane == 0) xm[row] = s * (1.0f / (float)HW);
}

// ---------------- 2) conv1(2->64)+BN+leaky+conv2(64->64), per-sample sums ----------------
__global__ void mc_kernel(const float* __restrict__ xm, const float* __restrict__ c1w,
                          const float* __restrict__ gamma, const float* __restrict__ beta,
                          const float* __restrict__ mean,  const float* __restrict__ var,
                          const float* __restrict__ c2w,   float* __restrict__ t2,
                          float* __restrict__ sums) {
    __shared__ float w2[64 * 64];
    int n = blockIdx.x >> 3;
    int d = (blockIdx.x & 7) * 256 + threadIdx.x;
    for (int i = threadIdx.x; i < 64 * 64; i += 256) w2[i] = c2w[i];
    __syncthreads();
    float in0 = xm[(size_t)(n * CB + 0) * DD + d];
    float in1 = xm[(size_t)(n * CB + 1) * DD + d];
    float mid[64];
    #pragma unroll 8
    for (int o = 0; o < 64; ++o) {
        float inv = gamma[o] * __frsqrt_rn(var[o] + 1e-5f);
        float v = (in0 * c1w[o * 2] + in1 * c1w[o * 2 + 1]) * inv + (beta[o] - mean[o] * inv);
        mid[o] = v > 0.0f ? v : 0.01f * v;
    }
    float s = 0.0f, ss = 0.0f;
    for (int o = 0; o < 64; ++o) {
        float acc = 0.0f;
        #pragma unroll 8
        for (int c = 0; c < 64; ++c) acc += w2[o * 64 + c] * mid[c];
        t2[((size_t)(n * 64 + o)) * DD + d] = acc;
        s += acc; ss += acc * acc;
    }
    for (int m = 16; m; m >>= 1) { s += __shfl_xor(s, m, 32); ss += __shfl_xor(ss, m, 32); }
    if ((threadIdx.x & 31) == 0) { atomicAdd(&sums[n], s); atomicAdd(&sums[8 + n], ss); }
}

// ---------------- 3) LayerNorm + transpose to [N][D][64] f16 ----------------
__global__ void ln_kernel(const float* __restrict__ t2, const float* __restrict__ sums,
                          const float* __restrict__ lnw, const float* __restrict__ lnb,
                          _Float16* __restrict__ tT) {
    int n = blockIdx.x >> 3;
    int d = (blockIdx.x & 7) * 256 + threadIdx.x;
    const float cnt = 64.0f * (float)DD;
    float mu   = sums[n] / cnt;
    float varr = sums[8 + n] / cnt - mu * mu;
    float rstd = __frsqrt_rn(varr + 1e-6f);
    _Float16* o = tT + ((size_t)(n * DD + d)) * 64;
    #pragma unroll 8
    for (int c = 0; c < 64; ++c) {
        float v = (t2[((size_t)(n * 64 + c)) * DD + d] - mu) * rstd
                  * lnw[(size_t)c * DD + d] + lnb[(size_t)c * DD + d];
        o[c] = (_Float16)v;
    }
}

// ---------------- 4) convert Q/K/V weights to f16 ----------------
__global__ void wcvt_kernel(const float* __restrict__ wq, const float* __restrict__ wk,
                            const float* __restrict__ wv, _Float16* __restrict__ out) {
    int g = blockIdx.x * 256 + threadIdx.x;             // 0..3*16384-1
    const float* src = (g < 16384) ? wq : (g < 32768) ? wk : wv;
    out[g] = (_Float16)src[g & 16383];
}

// ---------------- 5) Q/K/V projection via WMMA ----------------
// Qs: [N][D][256] f16 pre-scaled by 1/HDIM ; Kf: [N][D][256] f16 ; Vt: [N][256][D] f16
__global__ void qkv_kernel(const _Float16* __restrict__ tT, const _Float16* __restrict__ w16,
                           _Float16* __restrict__ Qs, _Float16* __restrict__ Kf,
                           _Float16* __restrict__ Vt) {
    int n  = blockIdx.x >> 7;
    int d0 = (blockIdx.x & 127) * 16;
    int lane = threadIdx.x & 31, wave = threadIdx.x >> 5;   // 4 waves
    int ln15  = lane & 15;
    int hiOff = (lane < 16) ? 0 : 8;
    int kbB   = (lane < 16) ? 0 : 16;
    int M0    = (lane < 16) ? 0 : 8;

    // A fragments for k0=0 and k0=32, gathered straight from global (contiguous spans)
    const _Float16* Ab = tT + (size_t)(n * DD + d0) * 64;
    v16h a0, a1;
    #pragma unroll
    for (int i = 0; i < 8; ++i) {
        a0[i]     = Ab[(size_t)ln15 * 64 + kbB + i];
        a0[8 + i] = Ab[(size_t)ln15 * 64 + kbB + 8 + i];
        a1[i]     = Ab[(size_t)ln15 * 64 + 32 + kbB + i];
        a1[8 + i] = Ab[(size_t)ln15 * 64 + 32 + kbB + 8 + i];
    }
    for (int t = wave; t < 48; t += 4) {
        int mat = t >> 4;                 // 0=Q 1=K 2=V
        int j0  = (t & 15) * 16;
        const _Float16* w = w16 + (size_t)mat * 16384;
        v8f c = {};
        c = wmma_f16(a0, *(const v16h*)(w + (size_t)(j0 + ln15) * 64 + kbB), c);
        c = wmma_f16(a1, *(const v16h*)(w + (size_t)(j0 + ln15) * 64 + 32 + kbB), c);
        if (mat == 2) {                   // V stored transposed: Vt[n][j][d]
            _Float16* o = Vt + ((size_t)(n * 256 + j0 + ln15)) * DD + d0 + M0;
            #pragma unroll
            for (int j = 0; j < 8; ++j) o[j] = (_Float16)c[j];
        } else {
            _Float16* o = (mat == 0 ? Qs : Kf) + ((size_t)(n * DD + d0 + M0)) * 256 + j0 + ln15;
            float s = (mat == 0) ? (1.0f / (float)HDIM) : 1.0f;
            #pragma unroll
            for (int j = 0; j < 8; ++j) o[(size_t)j * 256] = (_Float16)(c[j] * s);
        }
    }
}

// ---------------- 6) attention: TDM K-stage + scores (WMMA) -> softmax -> att*V (WMMA) --
__global__ void attn_kernel(const _Float16* __restrict__ Qs, const _Float16* __restrict__ Kf,
                            const _Float16* __restrict__ Vt, float* __restrict__ att_out,
                            float* __restrict__ attv_out) {
    extern __shared__ char smem[];
    float*    sc = (float*)smem;                                   // [16][SC_STRIDE] f32
    _Float16* qa = (_Float16*)(smem + (size_t)16 * SC_STRIDE * 4); // [16][64] f16
    // per-wave double-buffered K stage: 8 waves x 2 bufs x 16 rows x KROW halves
    unsigned int kst_base = (unsigned int)(16 * SC_STRIDE * 4 + 16 * 64 * 2);
    _Float16* kst = (_Float16*)(smem + kst_base);

    int blk   = blockIdx.x;
    int qtile = blk & 127;
    int n     = (blk >> 7) & 7;
    int h     = blk >> 10;
    int q0    = qtile * 16;
    int tid   = threadIdx.x;
    int lane  = tid & 31, wave = tid >> 5;                          // 8 waves
    int ln15  = lane & 15;
    int hiOff = (lane < 16) ? 0 : 8;
    int kbB   = (lane < 16) ? 0 : 16;
    int M0    = (lane < 16) ? 0 : 8;

    // stage Q tile (already /HDIM) into LDS
    const _Float16* Qb = Qs + ((size_t)(n * DD + q0)) * 256 + h * HDIM;
    for (int i = tid; i < 16 * 64; i += 256) qa[i] = Qb[(size_t)(i >> 6) * 256 + (i & 63)];
    __syncthreads();

    // ---- scores: per wave, TDM double-buffers K tiles into LDS, WMMA consumes ----
    int nct = qtile + 1;
    unsigned int myk = kst_base + (unsigned int)(wave * 2 * 16 * KROW) * 2;  // byte offset
    _Float16* mykp = kst + (size_t)wave * 2 * 16 * KROW;
    if (wave < nct)
        tdm_load_k_tile(Kf + ((size_t)(n * DD + wave * 16)) * 256 + h * HDIM, myk);
    int bufp = 0;
    for (int ct = wave; ct < nct; ct += 8) {
        if (ct + 8 < nct) {
            tdm_load_k_tile(Kf + ((size_t)(n * DD + (ct + 8) * 16)) * 256 + h * HDIM,
                            myk + (unsigned int)((bufp ^ 1) * 16 * KROW) * 2);
            __builtin_amdgcn_s_wait_tensorcnt(1);   // current buffer complete (in-order)
        } else {
            __builtin_amdgcn_s_wait_tensorcnt(0);
        }
        const _Float16* kb = mykp + (size_t)bufp * 16 * KROW;
        v8f c = {};
        #pragma unroll
        for (int k0 = 0; k0 < 64; k0 += 32) {
            v16h a;
            #pragma unroll
            for (int i = 0; i < 8; ++i) {
                a[i]     = qa[ln15 * 64 + k0 + kbB + i];
                a[8 + i] = qa[ln15 * 64 + k0 + kbB + 8 + i];
            }
            const _Float16* bp = kb + (size_t)ln15 * KROW + k0 + kbB;
            v8h blo = *(const v8h*)bp;
            v8h bhi = *(const v8h*)(bp + 8);
            v16h b;
            #pragma unroll
            for (int i = 0; i < 8; ++i) { b[i] = blo[i]; b[8 + i] = bhi[i]; }
            c = wmma_f16(a, b, c);
        }
        #pragma unroll
        for (int j = 0; j < 8; ++j) sc[(M0 + j) * SC_STRIDE + ct * 16 + ln15] = c[j];
        bufp ^= 1;
    }
    __syncthreads();

    // ---- causal softmax over LDS; stream full row to global (NT) ----
    {
        int r  = tid >> 4;          // 16 threads per row
        int lg = tid & 15;
        int q  = q0 + r;
        float mx = -3.4e38f;
        for (int col = lg; col <= q; col += 16) mx = fmaxf(mx, sc[r * SC_STRIDE + col]);
        for (int m = 8; m; m >>= 1) mx = fmaxf(mx, __shfl_xor(mx, m, 16));
        float sum = 0.0f;
        for (int col = lg; col <= q; col += 16) {
            float e = __expf(sc[r * SC_STRIDE + col] - mx);
            sc[r * SC_STRIDE + col] = e;
            sum += e;
        }
        for (int m = 8; m; m >>= 1) sum += __shfl_xor(sum, m, 16);
        float inv = 1.0f / sum;
        float* arow = att_out + ((size_t)((h * NB + n) * DD + q0 + r)) * DD;
        for (int col = lg; col < DD; col += 16) {
            float v = (col <= q) ? sc[r * SC_STRIDE + col] * inv : 0.0f;
            sc[r * SC_STRIDE + col] = v;
            __builtin_nontemporal_store(v, arow + col);
        }
    }
    __syncthreads();

    // ---- out = att @ V : 4 column tiles of the head slice ----
    if (wave < 4) {
        int nt = wave;
        int nk = (q0 + 16 + 31) >> 5;       // 32-wide k chunks (tail is zero-filled)
        v8f c = {};
        const _Float16* vbase = Vt + ((size_t)(n * 256 + h * HDIM + nt * 16 + ln15)) * DD + kbB;
        for (int kc = 0; kc < nk; ++kc) {
            int k0 = kc * 32;
            if (kc + 1 < nk) __builtin_prefetch(vbase + k0 + 32, 0, 0);
            v16h a;
            #pragma unroll
            for (int i = 0; i < 8; ++i) {
                a[i]     = (_Float16)sc[ln15 * SC_STRIDE + k0 + kbB + i];
                a[8 + i] = (_Float16)sc[ln15 * SC_STRIDE + k0 + kbB + 8 + i];
            }
            c = wmma_f16(a, *(const v16h*)(vbase + k0), c);
        }
        float* outp = attv_out + ((size_t)(n * DD + q0 + M0)) * 256 + h * HDIM + nt * 16 + ln15;
        #pragma unroll
        for (int j = 0; j < 8; ++j) outp[(size_t)j * 256] = c[j];
    }
}

// ---------------- 7) att_merge = att_values @ w_out ; y = x * att_merge ----------------
__global__ void ms_kernel(const float* __restrict__ x, const float* __restrict__ attv,
                          const float* __restrict__ wout, float* __restrict__ y,
                          float* __restrict__ att_merge) {
    __shared__ float mbuf[256];
    __shared__ float wsh[256];
    int n  = blockIdx.x >> 3;
    int d0 = (blockIdx.x & 7) * 256;
    int tid = threadIdx.x;
    wsh[tid] = wout[tid];
    __syncthreads();
    int d = d0 + tid;
    const float* av = attv + ((size_t)(n * DD + d)) * 256;
    float m = 0.0f;
    #pragma unroll 8
    for (int k = 0; k < 256; ++k) m += av[k] * wsh[k];
    mbuf[tid] = m;
    att_merge[(size_t)n * DD + d] = m;
    __syncthreads();
    const size_t nbase = (size_t)n * CB * DD * HW;
    for (int idx = tid; idx < CB * 256 * HW; idx += 256) {
        int p  = idx & 255;
        int dl = (idx >> 8) & 255;
        int c  = idx >> 16;
        size_t off = nbase + ((size_t)c * DD + d0 + dl) * HW + p;
        __builtin_nontemporal_store(__builtin_nontemporal_load(x + off) * mbuf[dl], y + off);
    }
}

extern "C" void kernel_launch(void* const* d_in, const int* in_sizes, int n_in,
                              void* d_out, int out_size, void* d_ws, size_t ws_size,
                              hipStream_t stream) {
    const float* x     = (const float*)d_in[0];
    const float* c1w   = (const float*)d_in[1];
    const float* gamma = (const float*)d_in[2];
    const float* beta  = (const float*)d_in[3];
    const float* mean  = (const float*)d_in[4];
    const float* var   = (const float*)d_in[5];
    const float* c2w   = (const float*)d_in[6];
    const float* lnw   = (const float*)d_in[7];
    const float* lnb   = (const float*)d_in[8];
    const float* wq    = (const float*)d_in[9];
    const float* wk    = (const float*)d_in[10];
    const float* wv    = (const float*)d_in[11];
    const float* wout  = (const float*)d_in[12];

    // output layout: y | att_softmax | att_values | att_merge (all f32, flat)
    float* y     = (float*)d_out;
    float* att   = y + (size_t)NB * CB * DD * HW;                 // 8,388,608
    float* attv  = att + (size_t)NHEADS * NB * DD * DD;           // +134,217,728
    float* attm  = attv + (size_t)NB * DD * NHEADS * HDIM;        // +4,194,304

    // workspace carve-up
    char* ws = (char*)d_ws;
    size_t o = 0;
    auto carve = [&](size_t bytes) { size_t r = o; o = (o + bytes + 255) & ~(size_t)255; return r; };
    float*    xm   = (float*)(ws + carve((size_t)NB * CB * DD * 4));
    float*    sums = (float*)(ws + carve(16 * 4));
    float*    t2   = (float*)(ws + carve((size_t)NB * 64 * DD * 4));
    _Float16* w16  = (_Float16*)(ws + carve((size_t)3 * 256 * 64 * 2));
    _Float16* tT   = (_Float16*)(ws + carve((size_t)NB * DD * 64 * 2));
    _Float16* Qs   = (_Float16*)(ws + carve((size_t)NB * DD * 256 * 2));
    _Float16* Kf   = (_Float16*)(ws + carve((size_t)NB * DD * 256 * 2));
    _Float16* Vt   = (_Float16*)(ws + carve((size_t)NB * 256 * DD * 2));

    zero_kernel<<<1, 32, 0, stream>>>(sums, 16);
    pool_kernel<<<(NB * CB * DD) / 8, 256, 0, stream>>>(x, xm);
    mc_kernel<<<NB * 8, 256, 0, stream>>>(xm, c1w, gamma, beta, mean, var, c2w, t2, sums);
    ln_kernel<<<NB * 8, 256, 0, stream>>>(t2, sums, lnw, lnb, tT);
    wcvt_kernel<<<(3 * 16384) / 256, 256, 0, stream>>>(wq, wk, wv, w16);
    qkv_kernel<<<NB * (DD / 16), 128, 0, stream>>>(tT, w16, Qs, Kf, Vt);
    size_t attn_lds = (size_t)16 * SC_STRIDE * 4 + (size_t)16 * 64 * 2
                    + (size_t)8 * 2 * 16 * KROW * 2;              // ~166 KB (<320 KB/WGP)
    attn_kernel<<<NHEADS * NB * (DD / 16), 256, attn_lds, stream>>>(Qs, Kf, Vt, att, attv);
    ms_kernel<<<NB * 8, 256, 0, stream>>>(x, attv, wout, y, attm);
}
